// Net_3006477107597
// MI455X (gfx1250) — matrix operand order repaired and checked by reading, not compiled, not run on hardware
//
#include <hip/hip_runtime.h>
#include <hip/hip_bf16.h>

// ---------------------------------------------------------------------------
// CDNA5 WMMA helpers (gfx1250, wave32). 16x16x32 f16 -> f32 accumulate.
// All fragment loads are UNCONDITIONAL: buffers are zero-padded instead of
// predicating loads (predication previously generated per-element exec-mask
// branches + serialized s_wait_dscnt).
// ---------------------------------------------------------------------------
typedef __attribute__((ext_vector_type(16))) _Float16 v16h;
typedef __attribute__((ext_vector_type(8)))  float    v8f;

union AF { v16h v; _Float16 h[16]; };
union CF { v8f  v; float    f[8];  };

__device__ inline v8f wmma16(v16h a, v16h b, v8f c) {
  // (neg_a, A, neg_b, B, c_mod, C, reuse_a, reuse_b)
  return __builtin_amdgcn_wmma_f32_16x16x32_f16(false, a, false, b,
                                                (short)0, c, false, false);
}

// A fragment: 16x32 f16 from row-major f32 [16 x ldx] (zero-padded).
// Lanes 0-15: M=lane, K = kbase..+15;
// lanes 16-31: same +16. Contiguous 16-float runs -> vectorizable LDS loads.
__device__ inline v16h load_A_f32(const float* X, int ldx, int kbase, int lane) {
  AF a;
  const int m = lane & 15;
  const int khalf = (lane < 16) ? 0 : 16;
  const float* p = X + m * ldx + kbase + khalf;
#pragma unroll
  for (int j = 0; j < 8; ++j) a.h[j] = (_Float16)p[j];
#pragma unroll
  for (int j = 0; j < 8; ++j) a.h[8 + j] = (_Float16)p[8 + j];
  return a.v;
}

// B fragment: 32x16 f16 from row-major f32 weight [K x ldw], all in-bounds.
// VGPR v: lanes 0-15 -> K=kbase+2v,2v+1 (N=lane); lanes 16-31 -> +16.
__device__ inline v16h load_B_f32(const float* W, int ldw, int ntile,
                                  int kbase, int lane) {
  AF b;
  const int n = ntile * 16 + (lane & 15);
  const int koff = (lane < 16) ? 0 : 16;
  const float* p = W + (size_t)(kbase + koff) * ldw + n;
#pragma unroll
  for (int j = 0; j < 16; ++j) b.h[j] = (_Float16)p[(size_t)j * ldw];
  return b.v;
}

// B fragment from pre-converted f16 LDS weight (layer 0, zero-padded K->64).
__device__ inline v16h load_B_h16(const _Float16* Wh, int ldw, int ntile,
                                  int kbase, int lane) {
  AF b;
  const int n = ntile * 16 + (lane & 15);
  const int koff = (lane < 16) ? 0 : 16;
  const _Float16* p = Wh + (kbase + koff) * ldw + n;
#pragma unroll
  for (int j = 0; j < 16; ++j) b.h[j] = p[j * ldw];
  return b.v;
}

// D store, full 16 rows (unconditional).
__device__ inline void store_D16(float* Y, int ldy, int ntile,
                                 const v8f& d, int lane) {
  CF c; c.v = d;
  const int n = ntile * 16 + (lane & 15);
  const int mbase = (lane < 16) ? 0 : 8;
#pragma unroll
  for (int r = 0; r < 8; ++r) Y[(mbase + r) * ldy + n] = c.f[r];
}

// D store, row 0 only (M=1 matmuls).
__device__ inline void store_D_row0(float* Y, int ntile, const v8f& d, int lane) {
  CF c; c.v = d;
  if (lane < 16) Y[ntile * 16 + lane] = c.f[0];
}

#define NN 16
#define NE 64
#define DD 128

// ---------------------------------------------------------------------------
// One block per tower: 4x (GCN -> ReLU -> SAGPool) -> lin -> log_softmax.
// Activations live in zero-padded LDS; W0 staged to f16 LDS once.
// ---------------------------------------------------------------------------
__global__ __launch_bounds__(256) void tower_kernel(
    const float* __restrict__ gp_x, const float* __restrict__ sp_x,
    const int* __restrict__ gp_src, const int* __restrict__ gp_dst,
    const int* __restrict__ sp_src, const int* __restrict__ sp_dst,
    const float* g_W0, const float* g_W, const float* g_b,
    const float* g_pWrel, const float* g_pbrel, const float* g_pWroot,
    const float* g_linW, const float* g_linb,
    const float* s_W0, const float* s_W, const float* s_b,
    const float* s_pWrel, const float* s_pbrel, const float* s_pWroot,
    const float* s_linW, const float* s_linb,
    float* __restrict__ ws_out)
{
  __shared__ float    xbuf[NN * DD];
  __shared__ float    hbuf[NN * DD];
  __shared__ _Float16 w0h[64 * DD];          // W0 zero-padded 45->64 rows, f16
  __shared__ float    emask[NE], enorm[NE];
  __shared__ int      esrc[NE], edst[NE];
  __shared__ float    dinv[NN], sc[NN], pvec[NN], qvec[NN], tnv[NN], keptf[NN];
  __shared__ int      perm[NN], pos[NN];
  __shared__ float    red0;

  const int tid  = (int)threadIdx.x;
  const int lane = tid & 31;
  const int wv   = tid >> 5;
  const int tower = (int)blockIdx.x;

  const float* X0     = tower ? sp_x   : gp_x;
  const int*   SRC0   = tower ? sp_src : gp_src;
  const int*   DST0   = tower ? sp_dst : gp_dst;
  const float* W0     = tower ? s_W0     : g_W0;
  const float* Ws     = tower ? s_W      : g_W;
  const float* Bs     = tower ? s_b      : g_b;
  const float* PWrel  = tower ? s_pWrel  : g_pWrel;
  const float* Pbrel  = tower ? s_pbrel  : g_pbrel;
  const float* PWroot = tower ? s_pWroot : g_pWroot;
  const float* LinW   = tower ? s_linW   : g_linW;
  const float* Linb   = tower ? s_linb   : g_linb;

  // x (16x45) -> zero-padded 16x128 LDS; stage W0 -> f16 LDS (pad K 45->64)
  for (int i = tid; i < NN * DD; i += 256) {
    const int r = i >> 7, c = i & (DD - 1);
    xbuf[i] = (c < 45) ? X0[r * 45 + c] : 0.f;
  }
  for (int i = tid; i < 64 * DD; i += 256) {
    const int r = i >> 7, c = i & (DD - 1);
    w0h[i] = (r < 45) ? (_Float16)W0[r * DD + c] : (_Float16)0.f;
  }
  for (int e = tid; e < NE; e += 256) {
    esrc[e] = SRC0[e]; edst[e] = DST0[e]; emask[e] = 1.f;
  }
  __syncthreads();

  int n = NN;
  for (int layer = 0; layer < 4; ++layer) {
    // ---- h = x @ W via WMMA: one 16-col tile per wave (8 waves x 16 = 128)
    {
      v8f acc = {};
      if (layer == 0) {
#pragma unroll
        for (int kb = 0; kb < 2; ++kb) {
          v16h a = load_A_f32(xbuf, DD, kb * 32, lane);
          v16h b = load_B_h16(w0h, DD, wv, kb * 32, lane);
          acc = wmma16(a, b, acc);
        }
      } else {
        const float* W = Ws + (size_t)(layer - 1) * DD * DD;
#pragma unroll
        for (int kb = 0; kb < 4; ++kb) {
          v16h a = load_A_f32(xbuf, DD, kb * 32, lane);
          v16h b = load_B_f32(W, DD, wv, kb * 32, lane);
          acc = wmma16(a, b, acc);
        }
      }
      store_D16(hbuf, DD, wv, acc, lane);
    }
    __syncthreads();

    // ---- degree / normalization (deterministic, branchless selects)
    if (tid < NN) {
      float dg = 1.f;
      for (int e = 0; e < NE; ++e) dg += (edst[e] == tid) ? emask[e] : 0.f;
      dinv[tid] = rsqrtf(dg);
    }
    __syncthreads();
    if (tid < NE) enorm[tid] = dinv[esrc[tid]] * dinv[edst[tid]] * emask[tid];
    __syncthreads();

    // ---- out = sum_e norm*h[src] + dinv^2*h + b  -> ReLU
    const float* bias = Bs + layer * DD;
    for (int i = tid; i < n * DD; i += 256) {
      const int r = i >> 7, c = i & (DD - 1);
      float acc = dinv[r] * dinv[r] * hbuf[r * DD + c] + bias[c];
      for (int e = 0; e < NE; ++e) {
        const float w = (edst[e] == r) ? enorm[e] : 0.f;
        acc += w * hbuf[esrc[e] * DD + c];
      }
      xbuf[r * DD + c] = fmaxf(acc, 0.f);
    }
    __syncthreads();

    // ---- SAGPool scorer: score = seg_sum(emask*p[src]->dst) + b_rel + q
    const float* wrel  = PWrel  + layer * DD;
    const float* wroot = PWroot + layer * DD;
    if (tid < n) {
      float p = 0.f, q = 0.f;
      for (int c = 0; c < DD; ++c) {
        const float xv = xbuf[tid * DD + c];
        p += xv * wrel[c]; q += xv * wroot[c];
      }
      pvec[tid] = p; qvec[tid] = q;
    }
    __syncthreads();
    if (tid < n) {
      float s = qvec[tid] + Pbrel[layer];
      for (int e = 0; e < NE; ++e) {
        const float w = (edst[e] == tid) ? emask[e] : 0.f;
        s += w * pvec[esrc[e]];
      }
      sc[tid] = s;
    }
    __syncthreads();

    // ---- stable top-k (k = ceil(n/2)), matches argsort(-score)
    const int k = (n + 1) >> 1;
    if (tid == 0) {
      unsigned used = 0u;
      for (int j = 0; j < k; ++j) {
        int best = 0; float bv = -3.0e38f;
        for (int i = 0; i < n; ++i)
          if (!((used >> i) & 1u) && sc[i] > bv) { bv = sc[i]; best = i; }
        used |= (1u << best);
        perm[j] = best;
        tnv[j]  = tanhf(sc[best]);
      }
      for (int i = 0; i < NN; ++i) { pos[i] = 0; keptf[i] = 0.f; }
      for (int j = 0; j < k; ++j) { pos[perm[j]] = j; keptf[perm[j]] = 1.f; }
    }
    __syncthreads();

    // ---- gather kept nodes, scale by tanh(score)
    for (int i = tid; i < k * DD; i += 256) {
      const int j = i >> 7, c = i & (DD - 1);
      hbuf[j * DD + c] = xbuf[perm[j] * DD + c] * tnv[j];
    }
    __syncthreads();
    for (int i = tid; i < k * DD; i += 256) xbuf[i] = hbuf[i];
    for (int i = k * DD + tid; i < NN * DD; i += 256) xbuf[i] = 0.f;  // keep pad
    if (tid < NE) {
      const int s0 = esrc[tid], d0 = edst[tid];
      const float em = emask[tid] * keptf[s0] * keptf[d0];
      esrc[tid] = pos[s0]; edst[tid] = pos[d0]; emask[tid] = em;
    }
    n = k;
    __syncthreads();
  }

  // ---- lin: (1x128) @ (128x128) + linb, then log_softmax over 128
  {
    v8f acc = {};
#pragma unroll
    for (int kb = 0; kb < 4; ++kb) {
      v16h a = load_A_f32(xbuf, DD, kb * 32, lane);   // rows 1..15 are zero
      v16h b = load_B_f32(LinW, DD, wv, kb * 32, lane);
      acc = wmma16(a, b, acc);
    }
    store_D_row0(hbuf, wv, acc, lane);
  }
  __syncthreads();
  if (tid < DD) hbuf[tid] += Linb[tid];
  __syncthreads();
  if (tid == 0) {
    float m = hbuf[0];
    for (int c = 1; c < DD; ++c) m = fmaxf(m, hbuf[c]);
    float s = 0.f;
    for (int c = 0; c < DD; ++c) s += expf(hbuf[c] - m);
    red0 = m + logf(s);
  }
  __syncthreads();
  if (tid < DD) ws_out[tower * DD + tid] = hbuf[tid] - red0;
}

// ---------------------------------------------------------------------------
// FC head: 3x [Linear -> LayerNorm -> ReLU] + final Linear. M=1, K=N=256.
// h padded to 16x256 (rows 1..15 zero) so A fragments are unconditional.
// ---------------------------------------------------------------------------
__global__ __launch_bounds__(256) void fc_kernel(
    const float* __restrict__ fcW, const float* __restrict__ fcb,
    const float* __restrict__ lnw, const float* __restrict__ lnb,
    const float* __restrict__ hin, float* __restrict__ out)
{
  __shared__ float hpad[16 * 256];
  __shared__ float t[256];
  __shared__ float red[2];

  const int tid  = (int)threadIdx.x;
  const int lane = tid & 31;
  const int wv   = tid >> 5;

  hpad[tid] = hin[tid];
  for (int i = 256 + tid; i < 16 * 256; i += 256) hpad[i] = 0.f;
  __builtin_prefetch(fcW + tid * 64, 0, 3);  // warm L2 for weights
  __syncthreads();

  for (int l = 0; l < 4; ++l) {
    const float* W = fcW + (size_t)l * 256 * 256;
    for (int nt = wv; nt < 16; nt += 8) {
      v8f acc = {};
#pragma unroll
      for (int kb = 0; kb < 8; ++kb) {
        v16h a = load_A_f32(hpad, 256, kb * 32, lane);
        v16h b = load_B_f32(W, 256, nt, kb * 32, lane);
        acc = wmma16(a, b, acc);
      }
      store_D_row0(t, nt, acc, lane);
    }
    __syncthreads();

    if (l == 3) {
      out[tid] = t[tid] + fcb[3 * 256 + tid];
      return;
    }

    t[tid] = t[tid] + fcb[l * 256 + tid];
    __syncthreads();
    if (tid == 0) {
      float mu = 0.f;
      for (int c = 0; c < 256; ++c) mu += t[c];
      mu *= (1.f / 256.f);
      float var = 0.f;
      for (int c = 0; c < 256; ++c) { const float d = t[c] - mu; var += d * d; }
      var *= (1.f / 256.f);
      red[0] = mu;
      red[1] = rsqrtf(var + 1e-5f);
    }
    __syncthreads();
    hpad[tid] = fmaxf((t[tid] - red[0]) * red[1] * lnw[l * 256 + tid]
                      + lnb[l * 256 + tid], 0.f);
    __syncthreads();
  }
}

// ---------------------------------------------------------------------------
extern "C" void kernel_launch(void* const* d_in, const int* in_sizes, int n_in,
                              void* d_out, int out_size, void* d_ws, size_t ws_size,
                              hipStream_t stream) {
  (void)in_sizes; (void)n_in; (void)out_size; (void)ws_size;

  const float* gp_x   = (const float*)d_in[0];
  const float* sp_x   = (const float*)d_in[1];
  const int*   gp_src = (const int*)  d_in[2];
  const int*   gp_dst = (const int*)  d_in[3];
  const int*   sp_src = (const int*)  d_in[4];
  const int*   sp_dst = (const int*)  d_in[5];
  const float* fcW    = (const float*)d_in[6];
  const float* fcb    = (const float*)d_in[7];
  const float* lnw    = (const float*)d_in[8];
  const float* lnb    = (const float*)d_in[9];
  const float* g_W0    = (const float*)d_in[10];
  const float* g_W     = (const float*)d_in[11];
  const float* g_b     = (const float*)d_in[12];
  const float* g_pWrel = (const float*)d_in[13];
  const float* g_pbrel = (const float*)d_in[14];
  const float* g_pWroot= (const float*)d_in[15];
  const float* g_linW  = (const float*)d_in[16];
  const float* g_linb  = (const float*)d_in[17];
  const float* s_W0    = (const float*)d_in[18];
  const float* s_W     = (const float*)d_in[19];
  const float* s_b     = (const float*)d_in[20];
  const float* s_pWrel = (const float*)d_in[21];
  const float* s_pbrel = (const float*)d_in[22];
  const float* s_pWroot= (const float*)d_in[23];
  const float* s_linW  = (const float*)d_in[24];
  const float* s_linb  = (const float*)d_in[25];

  float* ws = (float*)d_ws;  // [256]: tower0 logits | tower1 logits

  tower_kernel<<<2, 256, 0, stream>>>(
      gp_x, sp_x, gp_src, gp_dst, sp_src, sp_dst,
      g_W0, g_W, g_b, g_pWrel, g_pbrel, g_pWroot, g_linW, g_linb,
      s_W0, s_W, s_b, s_pWrel, s_pbrel, s_pWroot, s_linW, s_linb,
      ws);

  fc_kernel<<<1, 256, 0, stream>>>(fcW, fcb, lnw, lnb, ws, (float*)d_out);
}
